// InteractionBlock_34797825032818
// MI455X (gfx1250) — matrix operand order repaired and checked by reading, not compiled, not run on hardware
//
#include <hip/hip_runtime.h>
#include <hip/hip_bf16.h>

typedef __attribute__((ext_vector_type(16))) _Float16 v16h;
typedef __attribute__((ext_vector_type(8)))  float    v8f;

#define N_NODES   40000
#define N_EDGES   640000
#define HIDDEN    128
#define NFILT     128
#define NGAUSS    50
#define LN2F      0.69314718055994531f
#define PI_OVER_CUTOFF 0.31415926535897932f   // pi / 10

// ---------------------------------------------------------------------------
// WMMA fragment layout helpers (wave32, v_wmma_f32_16x16x32_f16)
//
// A (16x32 f16): lane l -> M = l&15, half = l>>4.
//   VGPR v in 0..3: K = half*8 + 2v, +1 ; VGPR v in 4..7: K = 16 + half*8 + 2(v-4), +1
// B (32x16 f16): lane l -> N = l&15, half = l>>4. VGPR v: same K mapping as A
//   (we pre-pack B into lane-major dwords: frag[tile*256 + lane*8 + v])
// C/D (16x16 f32): lane l -> N = l&15 ; VGPR r: M = r + 8*(l>>4)
// ---------------------------------------------------------------------------

__device__ __forceinline__ int a_kofs(int v, int half) {
  return (v < 4) ? (half * 8 + 2 * v) : (16 + half * 8 + 2 * (v - 4));
}

__device__ __forceinline__ v8f wmma_f16(v16h a, v16h b, v8f c) {
  return __builtin_amdgcn_wmma_f32_16x16x32_f16(
      /*neg_a=*/false, a, /*neg_b=*/false, b,
      /*c_mod=*/(short)0, c, /*reuse_a=*/false, /*reuse_b=*/false);
}

// Build an A fragment from a row-major f32 row (convert to f16, zero-pad K).
__device__ __forceinline__ v16h load_a_global_f32(const float* row, int kbase,
                                                  int kmax, int half) {
  v16h a;
#pragma unroll
  for (int v = 0; v < 8; ++v) {
    int k = kbase + a_kofs(v, half);
    float f0 = (k     < kmax) ? row[k]     : 0.f;
    float f1 = (k + 1 < kmax) ? row[k + 1] : 0.f;
    a[2 * v]     = (_Float16)f0;
    a[2 * v + 1] = (_Float16)f1;
  }
  return a;
}

// Build an A fragment from a row-major f16 row in LDS (stride 128).
__device__ __forceinline__ v16h load_a_lds_f16(const _Float16* row, int kbase,
                                               int half) {
  v16h a;
#pragma unroll
  for (int v = 0; v < 8; ++v) {
    int k = kbase + a_kofs(v, half);
    a[2 * v]     = row[k];
    a[2 * v + 1] = row[k + 1];
  }
  return a;
}

// Pre-packed B fragment: 8 consecutive dwords per lane (global memory).
__device__ __forceinline__ v16h load_b_frag(const unsigned* frags, int tile,
                                            int lane) {
  return *reinterpret_cast<const v16h*>(frags + (size_t)tile * 256 + lane * 8);
}

// Numerically stable shifted softplus.
__device__ __forceinline__ float sspf(float x) {
  return fmaxf(x, 0.f) + log1pf(__expf(-fabsf(x))) - LN2F;
}

// Generic pointer to __shared__ -> 32-bit LDS byte address (ISA 10.2: the LDS
// aperture maps generic addr[31:0] directly to the LDS offset).
__device__ __forceinline__ unsigned lds_addr_of(const void* p) {
  return (unsigned)(unsigned long long)p;
}

// Async copy 16B global -> LDS (gfx1250 GLOBAL_LOAD_ASYNC_TO_LDS_B128,
// tracked by ASYNCcnt). Emitted via inline asm for toolchain portability.
__device__ __forceinline__ void async_copy_b128(const void* gsrc,
                                                unsigned lds_byte_addr) {
  unsigned long long ga = (unsigned long long)gsrc;
  asm volatile("global_load_async_to_lds_b128 %0, %1, off"
               :
               : "v"(lds_byte_addr), "v"(ga)
               : "memory");
}

__device__ __forceinline__ void wait_asynccnt0() {
  asm volatile("s_wait_asynccnt 0x0" ::: "memory");
}

// ---------------------------------------------------------------------------
// Pack a weight matrix w[N][K] (row-major f32) into B fragments of w^T
// (so GEMM computes act @ w^T). K padded to ktiles*32 with zeros.
// Output dwords: out[(kt*8 + nt)*256 + lane*8 + v]
// ---------------------------------------------------------------------------
__global__ void pack_b_frag(const float* __restrict__ w, int N, int K,
                            int ktiles, unsigned* __restrict__ out) {
  int idx = blockIdx.x * blockDim.x + threadIdx.x;
  int total = ktiles * 8 * 256;
  if (idx >= total) return;
  int tile = idx >> 8;
  int r    = idx & 255;
  int lane = r & 31;
  int v    = r >> 5;
  int nt   = tile & 7;
  int kt   = tile >> 3;
  int n = nt * 16 + (lane & 15);
  int k = kt * 32 + ((v < 4) ? ((lane >> 4) * 8 + 2 * v) : (16 + (lane >> 4) * 8 + 2 * (v - 4)));
  float f0 = (n < N && k     < K) ? w[(size_t)n * K + k]     : 0.f;
  float f1 = (n < N && k + 1 < K) ? w[(size_t)n * K + k + 1] : 0.f;
  _Float16 h0 = (_Float16)f0, h1 = (_Float16)f1;
  unsigned u0 = __builtin_bit_cast(unsigned short, h0);
  unsigned u1 = __builtin_bit_cast(unsigned short, h1);
  out[(size_t)tile * 256 + lane * 8 + v] = (u1 << 16) | u0;
}

__global__ void zero_f4(float4* __restrict__ p, int n4) {
  int i = blockIdx.x * blockDim.x + threadIdx.x;
  if (i < n4) p[i] = make_float4(0.f, 0.f, 0.f, 0.f);
}

// ---------------------------------------------------------------------------
// h = x @ lin1_w^T   (one wave per 16-node strip)
// ---------------------------------------------------------------------------
__global__ void __launch_bounds__(128) node_lin1_kernel(
    const float* __restrict__ x, const unsigned* __restrict__ l1f,
    float* __restrict__ h) {
  int wave = threadIdx.x >> 5, lane = threadIdx.x & 31;
  int tile = blockIdx.x * 4 + wave;
  if (tile >= N_NODES / 16) return;
  int half = lane >> 4, col = lane & 15;
  const float* xrow = x + ((size_t)tile * 16 + col) * HIDDEN;

  v16h af[4];
#pragma unroll
  for (int kt = 0; kt < 4; ++kt)
    af[kt] = load_a_global_f32(xrow, kt * 32, HIDDEN, half);

#pragma unroll
  for (int nt = 0; nt < 8; ++nt) {
    v8f acc = {};
#pragma unroll
    for (int kt = 0; kt < 4; ++kt)
      acc = wmma_f16(af[kt], load_b_frag(l1f, kt * 8 + nt, lane), acc);
    int c = nt * 16 + col;
#pragma unroll
    for (int r = 0; r < 8; ++r) {
      int row = r + 8 * half;
      h[((size_t)tile * 16 + row) * NFILT + c] = acc[r];
    }
  }
}

// ---------------------------------------------------------------------------
// Edge kernel: filter MLP (two WMMA GEMMs + ssp) -> envelope -> gather h[src],
// multiply, scatter-add into agg[dst]. One wave per 16-edge tile.
// w2 fragments staged once per block into LDS via async-to-LDS DMA.
// ---------------------------------------------------------------------------
__global__ void __launch_bounds__(128) edge_kernel(
    const float* __restrict__ edge_attr, const float* __restrict__ edge_weight,
    const int* __restrict__ src, const int* __restrict__ dst,
    const unsigned* __restrict__ w1f, const unsigned* __restrict__ w2f,
    const float* __restrict__ b1, const float* __restrict__ b2,
    const float* __restrict__ h, float* __restrict__ agg) {
  __shared__ float    smem[4 * 16 * 128];  // 32 KB, 16x128 f32 per wave
  __shared__ unsigned lds_w2[32 * 256];    // 32 KB, staged w2 fragments

  int wave = threadIdx.x >> 5, lane = threadIdx.x & 31;
  int tile = blockIdx.x * 4 + wave;
  int half = lane >> 4, col = lane & 15;

  // ---- async stage of w2 fragment buffer into LDS (whole block) ----------
  {
    unsigned base = lds_addr_of(lds_w2);
#pragma unroll 4
    for (int i = threadIdx.x; i < 2048; i += 128) {  // 2048 x 16B = 32 KB
      async_copy_b128((const char*)w2f + (size_t)i * 16, base + i * 16);
    }
    wait_asynccnt0();
  }
  __syncthreads();

  if (tile >= N_EDGES / 16) return;

  float*    wsW = &smem[wave * 16 * 128];
  _Float16* wsT = reinterpret_cast<_Float16*>(wsW);  // first 4 KB as f16

  // ---- prefetch the h[src] gather rows for this tile behind the GEMMs ----
  {
    int s_pre = src[(size_t)tile * 16 + col];
    const char* hp = (const char*)(h + (size_t)s_pre * NFILT);
    __builtin_prefetch(hp + half * 256, 0, 0);        // two 256B halves
    __builtin_prefetch(hp + 128 + half * 256, 0, 0);  // cover full 512B row
  }

  const float* arow = edge_attr + ((size_t)tile * 16 + col) * NGAUSS;

  // ---- GEMM1: (16 x 50pad64) @ w1^T -> ssp -> f16 in LDS -----------------
  v16h a0 = load_a_global_f32(arow, 0,  NGAUSS, half);
  v16h a1 = load_a_global_f32(arow, 32, NGAUSS, half);
#pragma unroll
  for (int nt = 0; nt < 8; ++nt) {
    v8f acc = {};
    acc = wmma_f16(a0, load_b_frag(w1f, 0 * 8 + nt, lane), acc);
    acc = wmma_f16(a1, load_b_frag(w1f, 1 * 8 + nt, lane), acc);
    int c = nt * 16 + col;
    float bias = b1[c];
#pragma unroll
    for (int r = 0; r < 8; ++r) {
      int row = r + 8 * half;
      wsT[row * 128 + c] = (_Float16)sspf(acc[r] + bias);
    }
  }

  // ---- GEMM2: T1(16x128) @ w2^T (B from LDS) -> W f32 in LDS -------------
  v16h af[4];
#pragma unroll
  for (int kt = 0; kt < 4; ++kt)
    af[kt] = load_a_lds_f16(&wsT[col * 128], kt * 32, half);

#pragma unroll
  for (int nt = 0; nt < 8; ++nt) {
    v8f acc = {};
#pragma unroll
    for (int kt = 0; kt < 4; ++kt) {
      v16h b = *reinterpret_cast<const v16h*>(lds_w2 + (kt * 8 + nt) * 256 +
                                              lane * 8);
      acc = wmma_f16(af[kt], b, acc);
    }
    int c = nt * 16 + col;
    float bias = b2[c];
#pragma unroll
    for (int r = 0; r < 8; ++r) {
      int row = r + 8 * half;
      wsW[row * 128 + c] = acc[r] + bias;  // overwrites T1 (already consumed)
    }
  }

  // ---- envelope * gather * scatter-add (coalesced per edge row) ----------
  for (int row = 0; row < 16; ++row) {
    size_t e = (size_t)tile * 16 + row;
    int   s  = src[e];
    int   d  = dst[e];
    float env = 0.5f * (__cosf(edge_weight[e] * PI_OVER_CUTOFF) + 1.0f);
    const float* hrow  = h + (size_t)s * NFILT;
    float*       arow2 = agg + (size_t)d * NFILT;
#pragma unroll
    for (int it = 0; it < 4; ++it) {
      int f = lane + it * 32;
      float val = wsW[row * 128 + f] * env * hrow[f];
      __hip_atomic_fetch_add(&arow2[f], val, __ATOMIC_RELAXED,
                             __HIP_MEMORY_SCOPE_AGENT);
    }
  }
}

// ---------------------------------------------------------------------------
// out = ssp(agg @ lin2_w^T + lin2_b) @ lin_w^T + lin_b  (fused, one wave/tile)
// ---------------------------------------------------------------------------
__global__ void __launch_bounds__(128) node_out_kernel(
    const float* __restrict__ agg, const unsigned* __restrict__ l2f,
    const float* __restrict__ lin2_b, const unsigned* __restrict__ lwf,
    const float* __restrict__ lin_b, float* __restrict__ out) {
  __shared__ _Float16 sm[4 * 16 * 128];  // 16 KB, per-wave 16x128 f16
  int wave = threadIdx.x >> 5, lane = threadIdx.x & 31;
  int tile = blockIdx.x * 4 + wave;
  if (tile >= N_NODES / 16) return;
  int half = lane >> 4, col = lane & 15;
  _Float16* ws = &sm[wave * 16 * 128];

  const float* grow = agg + ((size_t)tile * 16 + col) * NFILT;
  v16h af[4];
#pragma unroll
  for (int kt = 0; kt < 4; ++kt)
    af[kt] = load_a_global_f32(grow, kt * 32, NFILT, half);

  // lin2 + bias + ssp -> f16 LDS
#pragma unroll
  for (int nt = 0; nt < 8; ++nt) {
    v8f acc = {};
#pragma unroll
    for (int kt = 0; kt < 4; ++kt)
      acc = wmma_f16(af[kt], load_b_frag(l2f, kt * 8 + nt, lane), acc);
    int c = nt * 16 + col;
    float bias = lin2_b[c];
#pragma unroll
    for (int r = 0; r < 8; ++r) {
      int row = r + 8 * half;
      ws[row * 128 + c] = (_Float16)sspf(acc[r] + bias);
    }
  }

  // output linear
  v16h af2[4];
#pragma unroll
  for (int kt = 0; kt < 4; ++kt)
    af2[kt] = load_a_lds_f16(&ws[col * 128], kt * 32, half);

#pragma unroll
  for (int nt = 0; nt < 8; ++nt) {
    v8f acc = {};
#pragma unroll
    for (int kt = 0; kt < 4; ++kt)
      acc = wmma_f16(af2[kt], load_b_frag(lwf, kt * 8 + nt, lane), acc);
    int c = nt * 16 + col;
    float bias = lin_b[c];
#pragma unroll
    for (int r = 0; r < 8; ++r) {
      int row = r + 8 * half;
      out[((size_t)tile * 16 + row) * HIDDEN + c] = acc[r] + bias;
    }
  }
}

// ---------------------------------------------------------------------------
extern "C" void kernel_launch(void* const* d_in, const int* in_sizes, int n_in,
                              void* d_out, int out_size, void* d_ws,
                              size_t ws_size, hipStream_t stream) {
  (void)in_sizes; (void)n_in; (void)out_size; (void)ws_size;
  const float* x       = (const float*)d_in[0];
  const int*   eidx    = (const int*)d_in[1];
  const int*   src     = eidx;
  const int*   dst     = eidx + N_EDGES;
  const float* ew      = (const float*)d_in[2];
  const float* ea      = (const float*)d_in[3];
  const float* mlp_w1  = (const float*)d_in[4];
  const float* mlp_b1  = (const float*)d_in[5];
  const float* mlp_w2  = (const float*)d_in[6];
  const float* mlp_b2  = (const float*)d_in[7];
  const float* lin1_w  = (const float*)d_in[8];
  const float* lin2_w  = (const float*)d_in[9];
  const float* lin2_b  = (const float*)d_in[10];
  const float* lin_w   = (const float*)d_in[11];
  const float* lin_b   = (const float*)d_in[12];
  float*       out     = (float*)d_out;

  char* ws = (char*)d_ws;
  unsigned* w1f = (unsigned*)(ws + 0);        // 2 ktiles * 8 * 256 dw = 16 KB
  unsigned* w2f = (unsigned*)(ws + 16384);    // 4 * 8 * 256 dw = 32 KB
  unsigned* l1f = (unsigned*)(ws + 49152);    // 32 KB
  unsigned* l2f = (unsigned*)(ws + 81920);    // 32 KB
  unsigned* lwf = (unsigned*)(ws + 114688);   // 32 KB
  float* h   = (float*)(ws + 147456);                                 // 20.48 MB
  float* agg = (float*)(ws + 147456 + (size_t)N_NODES * NFILT * 4);   // 20.48 MB

  // Pack weights into B-fragment layout (w^T, K zero-padded).
  pack_b_frag<<<16, 256, 0, stream>>>(mlp_w1, NFILT, NGAUSS, 2, w1f);
  pack_b_frag<<<32, 256, 0, stream>>>(mlp_w2, NFILT, NFILT, 4, w2f);
  pack_b_frag<<<32, 256, 0, stream>>>(lin1_w, NFILT, HIDDEN, 4, l1f);
  pack_b_frag<<<32, 256, 0, stream>>>(lin2_w, HIDDEN, NFILT, 4, l2f);
  pack_b_frag<<<32, 256, 0, stream>>>(lin_w,  HIDDEN, HIDDEN, 4, lwf);

  // Zero the scatter accumulator every launch (graph-replay safe).
  zero_f4<<<(N_NODES * NFILT / 4 + 255) / 256, 256, 0, stream>>>(
      (float4*)agg, N_NODES * NFILT / 4);

  // h = x @ lin1_w^T
  node_lin1_kernel<<<N_NODES / 16 / 4, 128, 0, stream>>>(x, l1f, h);

  // Edge filter MLP + CFConv gather/scatter.
  edge_kernel<<<N_EDGES / 16 / 4, 128, 0, stream>>>(
      ea, ew, src, dst, w1f, w2f, mlp_b1, mlp_b2, h, agg);

  // out = ssp(agg @ lin2^T + b) @ lin^T + b
  node_out_kernel<<<N_NODES / 16 / 4, 128, 0, stream>>>(
      agg, l2f, lin2_b, lwf, lin_b, out);
}
